// GNN_Layer_34832184770734
// MI455X (gfx1250) — compile-verified, matrix operand order checked by reading, not run on hardware
//
#include <hip/hip_runtime.h>

#define DN 128
#define G3 384
#define DH 512
#define NNODES 20000
#define NEDGES 320000
#define LDA 136   // f16 halves pitch for 16x128 A tiles (bank-conflict-free)
#define LDM 132   // f32 pitch for 16x128 fp32 tiles
#define LDG 388   // f32 pitch for 16x384 gh tile

static_assert(NNODES % 16 == 0, "node tiles exact");
static_assert(NEDGES % 16 == 0, "edge tiles exact");

typedef __attribute__((ext_vector_type(16))) _Float16 v16h;
typedef __attribute__((ext_vector_type(8)))  _Float16 v8h;
typedef __attribute__((ext_vector_type(8)))  float    v8f;

// ---------- WMMA fragment helpers (wave32, 16x16x32 f16 -> f32) ----------
// A: 16x32 f16. lane(0-15)=row M, lanes 16-31 same M with K+8/K+24 halves.
__device__ __forceinline__ v16h a_frag(const _Float16* at, int k0, int lane) {
  const _Float16* p = at + (lane & 15) * LDA + k0 + ((lane >> 4) << 3);
  v8h lo = *(const v8h*)p;         // K = base .. base+7
  v8h hi = *(const v8h*)(p + 16);  // K = base+16 .. base+23
  v16h f;
#pragma unroll
  for (int i = 0; i < 8; ++i) { f[i] = lo[i]; f[i + 8] = hi[i]; }
  return f;
}

// Hoist all 4 K-chunk A fragments of a 16x128 tile (reused across out tiles).
__device__ __forceinline__ void load_af(const _Float16* at, int lane, v16h* af) {
#pragma unroll
  for (int c = 0; c < 4; ++c) af[c] = a_frag(at, c << 5, lane);
}

// B fragments pre-linearized: [(tile*4+chunk)*512 + lane*16 + e] halves.
__device__ __forceinline__ v16h b_frag(const _Float16* w, int tile, int chunk, int lane) {
  return *(const v16h*)(w + (((tile << 2) + chunk) << 9) + (lane << 4));
}

// One 16x16 output tile: K=128 accumulated over 4 WMMAs, A preloaded.
__device__ __forceinline__ v8f gemm_pre(const v16h* af, const _Float16* w,
                                        int tile, int lane, v8f acc) {
#pragma unroll
  for (int c = 0; c < 4; ++c)
    acc = __builtin_amdgcn_wmma_f32_16x16x32_f16(
        false, af[c], false, b_frag(w, tile, c, lane), (short)0, acc, false, false);
  return acc;
}

__device__ __forceinline__ void copy_lds(_Float16* dst, const _Float16* src,
                                         int halves, int tid, int nth) {
  uint4* d = (uint4*)dst; const uint4* s = (const uint4*)src;
  int vecs = halves >> 3;
  for (int i = tid; i < vecs; i += nth) d[i] = s[i];
}

// Stage 16 rows x 128 cols fp32 -> f16 LDS tile (+ optional fp32 copy).
// Exactly one iteration per thread (256 threads). Prefetch next tile rows.
__device__ __forceinline__ void stage_a16(_Float16* at, float* a32,
                                          const float* __restrict__ M,
                                          int row0, int tid,
                                          int pfrow0, int nmax) {
  int r = tid >> 4, c0 = (tid & 15) << 3;
  const float4* p = (const float4*)(M + (long)(row0 + r) * DN + c0);
  float4 u = p[0], v = p[1];
  v8h h;
  h[0] = (_Float16)u.x; h[1] = (_Float16)u.y; h[2] = (_Float16)u.z; h[3] = (_Float16)u.w;
  h[4] = (_Float16)v.x; h[5] = (_Float16)v.y; h[6] = (_Float16)v.z; h[7] = (_Float16)v.w;
  *(v8h*)(at + r * LDA + c0) = h;
  if (a32) {
    *(float4*)(a32 + r * LDM + c0) = u;
    *(float4*)(a32 + r * LDM + c0 + 4) = v;
  }
  if (pfrow0 < nmax)
    __builtin_prefetch(M + (long)(pfrow0 + r) * DN + c0, 0, 1);
}

// Store one 16x16 f32 tile (C/D layout: reg r -> row rbase+r, lane -> col).
__device__ __forceinline__ void store_tile(float* __restrict__ out, int ld, int row0,
                                           int n0, int lane, const v8f acc, float bias) {
  int col = n0 + (lane & 15);
  int base = (row0 + ((lane >> 4) << 3)) * ld + col;
#pragma unroll
  for (int r = 0; r < 8; ++r) out[base + r * ld] = acc[r] + bias;
}

// ---------- weight pre-conversion: fp32 [out x 128] -> f16 B-fragment layout ----
// Global tile map: [0,8)=W_hu [8,16)=W_hw [16,40)=w_ih [40,48)=W_e [48,80)=W_attn [80,104)=w_hh
__global__ void k_wprep(const float* __restrict__ W_hu, const float* __restrict__ W_hw,
                        const float* __restrict__ w_ih, const float* __restrict__ W_e,
                        const float* __restrict__ W_attn, const float* __restrict__ w_hh,
                        _Float16* __restrict__ wfrag) {
  const int total = 104 * 2048;
  for (int idx = blockIdx.x * blockDim.x + threadIdx.x; idx < total;
       idx += gridDim.x * blockDim.x) {
    int e = idx & 15;
    int lane = (idx >> 4) & 31;
    int chunk = (idx >> 9) & 3;
    int tile = idx >> 11;
    const float* W; int t;
    if (tile < 8)       { W = W_hu;   t = tile; }
    else if (tile < 16) { W = W_hw;   t = tile - 8; }
    else if (tile < 40) { W = w_ih;   t = tile - 16; }
    else if (tile < 48) { W = W_e;    t = tile - 40; }
    else if (tile < 80) { W = W_attn; t = tile - 48; }
    else                { W = w_hh;   t = tile - 80; }
    int n = (t << 4) + (lane & 15);
    int k = (chunk << 5) + ((lane >> 4) << 3) + ((e & 8) << 1) + (e & 7);
    wfrag[idx] = (_Float16)W[n * DN + k];
  }
}

__global__ void k_init(float* __restrict__ seg_sum, float* __restrict__ agg) {
  int i = blockIdx.x * blockDim.x + threadIdx.x;
  const int nseg = NNODES * DH;
  if (i < nseg) seg_sum[i] = 0.f;
  else if (i < nseg + NNODES * DN) agg[i - nseg] = 0.f;
}

// ---------- node pre-pass: xu = x@W_hu^T, xw = x@W_hw^T, gi = x@w_ih^T + b_ih ----
__global__ void __launch_bounds__(256) k_node_pre(
    const float* __restrict__ x, const _Float16* __restrict__ wfrag_node,
    const float* __restrict__ b_ih,
    float* __restrict__ xu, float* __restrict__ xw, float* __restrict__ gi) {
  extern __shared__ char smem[];
  _Float16* wlds = (_Float16*)smem;                  // 40 tiles = 160KB
  _Float16* at = (_Float16*)(smem + 40 * 2048 * 2);  // 16xLDA halves
  const int tid = threadIdx.x;
  const int lane = tid & 31, wave = tid >> 5;
  copy_lds(wlds, wfrag_node, 40 * 2048, tid, blockDim.x);
  const int ntiles = NNODES / 16;
  for (int t = blockIdx.x; t < ntiles; t += gridDim.x) {
    int row0 = t << 4;
    __syncthreads();
    stage_a16(at, nullptr, x, row0, tid, (t + gridDim.x) << 4, NNODES);
    __syncthreads();
    v16h af[4];
    load_af(at, lane, af);
    {  // j=0: xu, W_hu tile `wave`
      v8f acc = {};
      acc = gemm_pre(af, wlds, wave, lane, acc);
      store_tile(xu, DN, row0, wave << 4, lane, acc, 0.f);
    }
    {  // j=1: xw, W_hw tile `8+wave`
      v8f acc = {};
      acc = gemm_pre(af, wlds, 8 + wave, lane, acc);
      store_tile(xw, DN, row0, wave << 4, lane, acc, 0.f);
    }
#pragma unroll
    for (int j = 0; j < 3; ++j) {  // gi, w_ih tiles 16..39
      int tt = wave + (j << 3);
      v8f acc = {};
      acc = gemm_pre(af, wlds, 16 + tt, lane, acc);
      int col = (tt << 4) + (lane & 15);
      store_tile(gi, G3, row0, tt << 4, lane, acc, b_ih[col]);
    }
  }
}

// ---------- edge pass 1: messages + seg_sum[src] += exp(logits) ----------
__global__ void __launch_bounds__(256) k_edge_msg(
    const float* __restrict__ edge_attr, const int* __restrict__ src,
    const int* __restrict__ tgt, const _Float16* __restrict__ wfrag_edge,
    const float* __restrict__ xu, const float* __restrict__ xw,
    float* __restrict__ msg_out, float* __restrict__ seg_sum) {
  extern __shared__ char smem[];
  _Float16* wlds = (_Float16*)smem;                          // 40 tiles (W_e 0..7, W_attn 8..39)
  _Float16* at = (_Float16*)(smem + 40 * 2048 * 2);          // edge_attr tile
  _Float16* mt = (_Float16*)(smem + 40 * 2048 * 2 + 16 * LDA * 2);  // message tile (f16)
  int* sidx = (int*)(smem + 40 * 2048 * 2 + 2 * 16 * LDA * 2);
  int* tidx = sidx + 16;
  const int tid = threadIdx.x;
  const int lane = tid & 31, wave = tid >> 5;
  copy_lds(wlds, wfrag_edge, 40 * 2048, tid, blockDim.x);
  const int ntiles = NEDGES / 16;
  for (int t = blockIdx.x; t < ntiles; t += gridDim.x) {
    int e0 = t << 4;
    __syncthreads();
    stage_a16(at, nullptr, edge_attr, e0, tid, (t + gridDim.x) << 4, NEDGES);
    if (tid < 16) {
      sidx[tid] = src[e0 + tid];
      tidx[tid] = tgt[e0 + tid];
    }
    __syncthreads();
    {  // messages: W_e tile `wave`
      v16h af[4];
      load_af(at, lane, af);
      v8f acc = {};
      acc = gemm_pre(af, wlds, wave, lane, acc);
      int col = (wave << 4) + (lane & 15);
      int rbase = (lane >> 4) << 3;
#pragma unroll
      for (int r = 0; r < 8; ++r) {
        int rl = rbase + r;
        float v = acc[r] + xu[sidx[rl] * DN + col] + xw[tidx[rl] * DN + col];
        v = v > 0.f ? v : 0.01f * v;  // LeakyReLU
        mt[rl * LDA + col] = (_Float16)v;
        msg_out[(e0 + rl) * DN + col] = v;
      }
    }
    __syncthreads();
    v16h mf[4];
    load_af(mt, lane, mf);
    const int rbase = (lane >> 4) << 3;
#pragma unroll
    for (int j = 0; j < 4; ++j) {  // attention logits -> exp -> seg_sum
      int a = wave + (j << 3);     // attn col-tile 0..31
      v8f acc = {};
      acc = gemm_pre(mf, wlds, 8 + a, lane, acc);
      int col = (a << 4) + (lane & 15);  // 0..511 = h*128+d
#pragma unroll
      for (int r = 0; r < 8; ++r)
        unsafeAtomicAdd(&seg_sum[sidx[rbase + r] * DH + col], __expf(acc[r]));
    }
  }
}

// ---------- edge pass 2: agg[tgt] += m * (1/H) * sum_h exp(l)/seg_sum[src] ----
__global__ void __launch_bounds__(256) k_edge_agg(
    const float* __restrict__ msg, const int* __restrict__ src,
    const int* __restrict__ tgt, const _Float16* __restrict__ wfrag_attn,
    const float* __restrict__ seg_sum, float* __restrict__ agg) {
  extern __shared__ char smem[];
  _Float16* wlds = (_Float16*)smem;                         // 32 W_attn tiles = 128KB
  _Float16* mt = (_Float16*)(smem + 32 * 2048 * 2);
  float* m32 = (float*)(smem + 32 * 2048 * 2 + 16 * LDA * 2);
  int* sidx = (int*)(smem + 32 * 2048 * 2 + 16 * LDA * 2 + 16 * LDM * 4);
  int* tidx = sidx + 16;
  const int tid = threadIdx.x;
  const int lane = tid & 31, wave = tid >> 5;
  copy_lds(wlds, wfrag_attn, 32 * 2048, tid, blockDim.x);
  const int ntiles = NEDGES / 16;
  for (int t = blockIdx.x; t < ntiles; t += gridDim.x) {
    int e0 = t << 4;
    __syncthreads();
    stage_a16(mt, m32, msg, e0, tid, (t + gridDim.x) << 4, NEDGES);
    if (tid < 16) {
      sidx[tid] = src[e0 + tid];
      tidx[tid] = tgt[e0 + tid];
    }
    __syncthreads();
    v16h mf[4];
    load_af(mt, lane, mf);
    float ratio[8];
#pragma unroll
    for (int r = 0; r < 8; ++r) ratio[r] = 0.f;
    const int rbase = (lane >> 4) << 3;
#pragma unroll
    for (int h = 0; h < 4; ++h) {
      int a = (h << 3) + wave;  // cols 128h + 16*wave + N
      v8f acc = {};
      acc = gemm_pre(mf, wlds, a, lane, acc);
      int col = (a << 4) + (lane & 15);
#pragma unroll
      for (int r = 0; r < 8; ++r) {
        float denom = seg_sum[sidx[rbase + r] * DH + col];
        ratio[r] += __expf(acc[r]) / denom;
      }
    }
    int dcol = (wave << 4) + (lane & 15);
#pragma unroll
    for (int r = 0; r < 8; ++r) {
      int rl = rbase + r;
      float v = m32[rl * LDM + dcol] * ratio[r] * 0.25f;
      unsafeAtomicAdd(&agg[tidx[rl] * DN + dcol], v);
    }
  }
}

// ---------- GRU: gh = agg@w_hh^T + b_hh; gates; embeddings ----------
__global__ void __launch_bounds__(256) k_gru(
    const float* __restrict__ agg, const _Float16* __restrict__ wfrag_hh,
    const float* __restrict__ b_hh, const float* __restrict__ gi,
    float* __restrict__ emb) {
  extern __shared__ char smem[];
  _Float16* wlds = (_Float16*)smem;                  // 24 tiles = 96KB
  _Float16* at = (_Float16*)(smem + 24 * 2048 * 2);
  float* a32 = (float*)(smem + 24 * 2048 * 2 + 16 * LDA * 2);
  float* gh = (float*)(smem + 24 * 2048 * 2 + 16 * LDA * 2 + 16 * LDM * 4);
  const int tid = threadIdx.x;
  const int lane = tid & 31, wave = tid >> 5;
  copy_lds(wlds, wfrag_hh, 24 * 2048, tid, blockDim.x);
  const int ntiles = NNODES / 16;
  for (int t = blockIdx.x; t < ntiles; t += gridDim.x) {
    int row0 = t << 4;
    __syncthreads();
    stage_a16(at, a32, agg, row0, tid, (t + gridDim.x) << 4, NNODES);
    __syncthreads();
    v16h af[4];
    load_af(at, lane, af);
#pragma unroll
    for (int j = 0; j < 3; ++j) {
      int tt = wave + (j << 3);  // 0..23
      v8f acc = {};
      acc = gemm_pre(af, wlds, tt, lane, acc);
      int col = (tt << 4) + (lane & 15);  // 0..383
      float bias = b_hh[col];
      int base = ((lane >> 4) << 3) * LDG + col;
#pragma unroll
      for (int r = 0; r < 8; ++r) gh[base + r * LDG] = acc[r] + bias;
    }
    __syncthreads();
    {
      int r = tid >> 4, c0 = (tid & 15) << 3;
      const float* gir = gi + row0 * G3 + r * G3;
#pragma unroll
      for (int q = 0; q < 8; ++q) {
        int c = c0 + q;
        float i_r = gir[c], i_z = gir[DN + c], i_n = gir[2 * DN + c];
        float h_r = gh[r * LDG + c], h_z = gh[r * LDG + DN + c], h_n = gh[r * LDG + 2 * DN + c];
        float a = a32[r * LDM + c];
        float rr = 1.f / (1.f + __expf(-(i_r + h_r)));
        float zz = 1.f / (1.f + __expf(-(i_z + h_z)));
        float pre = fminf(fmaxf(i_n + rr * h_n, -15.f), 15.f);
        float e2 = __expf(2.f * pre);
        float nn = (e2 - 1.f) / (e2 + 1.f);  // tanh
        emb[(row0 + r) * DN + c] = (1.f - zz) * nn + zz * a;
      }
    }
  }
}

extern "C" void kernel_launch(void* const* d_in, const int* in_sizes, int n_in,
                              void* d_out, int out_size, void* d_ws, size_t ws_size,
                              hipStream_t stream) {
  (void)in_sizes; (void)n_in; (void)out_size; (void)ws_size;
  const float* x        = (const float*)d_in[0];
  const int* edge_index = (const int*)d_in[1];
  const float* edge_attr= (const float*)d_in[2];
  // d_in[3] = batch: only repacks in the reference; node-order output equals it.
  const float* W_e   = (const float*)d_in[4];
  const float* W_hu  = (const float*)d_in[5];
  const float* W_hw  = (const float*)d_in[6];
  const float* W_attn= (const float*)d_in[7];
  const float* w_ih  = (const float*)d_in[8];
  const float* w_hh  = (const float*)d_in[9];
  const float* b_ih  = (const float*)d_in[10];
  const float* b_hh  = (const float*)d_in[11];
  const int* src = edge_index;
  const int* tgt = edge_index + NEDGES;

  char* ws = (char*)d_ws;
  size_t off = 0;
  float* xu      = (float*)(ws + off); off += (size_t)NNODES * DN * 4;
  float* xw      = (float*)(ws + off); off += (size_t)NNODES * DN * 4;
  float* gi      = (float*)(ws + off); off += (size_t)NNODES * G3 * 4;
  float* seg_sum = (float*)(ws + off); off += (size_t)NNODES * DH * 4;
  float* agg     = (float*)(ws + off); off += (size_t)NNODES * DN * 4;
  _Float16* wfrag = (_Float16*)(ws + off);  // 104 tiles * 2048 halves = 416KB

  float* emb = (float*)d_out;
  float* msg = emb + (size_t)NNODES * DN;

  k_wprep<<<832, 256, 0, stream>>>(W_hu, W_hw, w_ih, W_e, W_attn, w_hh, wfrag);
  int initN = NNODES * (DH + DN);
  k_init<<<(initN + 255) / 256, 256, 0, stream>>>(seg_sum, agg);
  k_node_pre<<<640, 256, 40 * 2048 * 2 + 16 * LDA * 2, stream>>>(
      x, wfrag, b_ih, xu, xw, gi);
  k_edge_msg<<<1024, 256, 40 * 2048 * 2 + 2 * 16 * LDA * 2 + 128, stream>>>(
      edge_attr, src, tgt, wfrag + 40 * 2048, xu, xw, msg, seg_sum);
  k_edge_agg<<<1024, 256, 32 * 2048 * 2 + 16 * LDA * 2 + 16 * LDM * 4 + 128, stream>>>(
      msg, src, tgt, wfrag + 48 * 2048, seg_sum, agg);
  k_gru<<<640, 256, 24 * 2048 * 2 + 16 * LDA * 2 + 16 * LDM * 4 + 16 * LDG * 4, stream>>>(
      agg, wfrag + 80 * 2048, b_hh, gi, emb);
}
